// InstanceMatteHead_73701638799744
// MI455X (gfx1250) — compile-verified, matrix operand order checked by reading, not run on hardware
//
#include <hip/hip_runtime.h>
#include <hip/hip_bf16.h>

typedef __attribute__((ext_vector_type(16))) __bf16 v16bf;
typedef __attribute__((ext_vector_type(8)))  float  v8f;

#define NB    4        // b*nf
#define NI    10       // instances
#define CIN   32
#define HID   32
#define KOUT  8
#define KIN   64
#define IMG   256
#define TW    16       // tile edge
#define KTOT  288      // 9 taps * 32 channels
#define XSTR  34       // bf16 per (y,x) slot in x tile (32 used + pad)
#define HSTR  33       // f32 per pixel in hx tile (32 used + pad)
#define LEAKY 0.2f
#define EPS_BN 1e-5f

// ---------------------------------------------------------------------------
// Kernel 0: fold BN, build a[n][h][yc][xc] table, repack w_x -> bf16 (k = tap*32+c)
// ---------------------------------------------------------------------------
__global__ void setup_kernel(const float* __restrict__ emb_w,
                             const float* __restrict__ klin_w,
                             const float* __restrict__ conv1_w,
                             const float* __restrict__ bn_gamma,
                             const float* __restrict__ bn_beta,
                             const float* __restrict__ bn_mean,
                             const float* __restrict__ bn_var,
                             const float* __restrict__ conv2_w,
                             const float* __restrict__ conv2_b,
                             float* __restrict__ ws_a,
                             float* __restrict__ ws_inv,
                             float* __restrict__ ws_v,
                             float* __restrict__ ws_bias,
                             unsigned short* __restrict__ ws_wt_raw) {
  __shared__ float s_instk[NI * KOUT];
  __shared__ float s_inv[HID];
  __shared__ float s_bp[HID];
  const int tid = threadIdx.x;

  if (tid < NI * KOUT) {                     // inst_k = emb_w @ klin_w^T
    int n = tid / KOUT, k = tid - n * KOUT;
    float s = 0.f;
    for (int i = 0; i < KIN; ++i) s += emb_w[n * KIN + i] * klin_w[k * KIN + i];
    s_instk[tid] = s;
  }
  if (tid < HID) {                           // BN fold
    float iv = bn_gamma[tid] * rsqrtf(bn_var[tid] + EPS_BN);
    s_inv[tid] = iv;
    s_bp[tid]  = bn_beta[tid] - bn_mean[tid] * iv;
    ws_inv[tid] = iv;
    ws_v[tid]   = conv2_w[tid];              // conv2_w[0,:,0,0]
  }
  if (tid == 0) ws_bias[0] = conv2_b[0];
  __syncthreads();

  // a[n][h][cls] = (sum_k inst_k[n,k] * S9[h,k,cls]) * inv[h] + beta'
  for (int idx = tid; idx < NI * HID * 9; idx += blockDim.x) {
    int cls = idx % 9, xc = cls % 3, yc = cls / 3;
    int h = (idx / 9) % HID;
    int n = idx / (9 * HID);
    int dy0 = (yc == 0) ? 1 : 0, dy1 = (yc == 2) ? 1 : 2;
    int dx0 = (xc == 0) ? 1 : 0, dx1 = (xc == 2) ? 1 : 2;
    float acc = 0.f;
    for (int k = 0; k < KOUT; ++k) {
      const float* wk = conv1_w + (h * (CIN + KOUT) + (CIN + k)) * 9;
      float s9 = 0.f;
      for (int dy = dy0; dy <= dy1; ++dy)
        for (int dx = dx0; dx <= dx1; ++dx) s9 += wk[dy * 3 + dx];
      acc += s_instk[n * KOUT + k] * s9;
    }
    ws_a[idx] = acc * s_inv[h] + s_bp[h];
  }

  // w_x repack: Wt[h][tap*32 + c] = bf16(conv1_w[h][c][tap])
  __bf16* ws_wt = (__bf16*)ws_wt_raw;
  for (int idx = tid; idx < HID * KTOT; idx += blockDim.x) {
    int h = idx / KTOT, k = idx - h * KTOT;
    int tap = k >> 5, c = k & 31;
    ws_wt[idx] = (__bf16)conv1_w[(h * (CIN + KOUT) + c) * 9 + tap];
  }
}

// ---------------------------------------------------------------------------
// Kernel 1: per-(b,n) activity flags from masks (OR-reduce of 64K ints)
// ---------------------------------------------------------------------------
__global__ void active_kernel(const int* __restrict__ masks,
                              int* __restrict__ flags) {
  __shared__ int red[256];
  const int bn = blockIdx.x;
  const int* p = masks + (size_t)bn * IMG * IMG;
  int acc = 0;
  for (int i = threadIdx.x; i < IMG * IMG; i += 256) acc |= p[i];
  red[threadIdx.x] = acc;
  __syncthreads();
  for (int s = 128; s > 0; s >>= 1) {
    if (threadIdx.x < s) red[threadIdx.x] |= red[threadIdx.x + s];
    __syncthreads();
  }
  if (threadIdx.x == 0) flags[bn] = (red[0] != 0);
}

// ---------------------------------------------------------------------------
// Kernel 2: implicit-GEMM 3x3 conv via WMMA bf16 + fused BN/leaky/dot epilogue
// ---------------------------------------------------------------------------
__global__ __launch_bounds__(256) void matte_main(
    const float* __restrict__ x,
    const float* __restrict__ ws_a,
    const float* __restrict__ ws_inv,
    const float* __restrict__ ws_v,
    const float* __restrict__ ws_bias,
    const int* __restrict__ flags,
    const unsigned short* __restrict__ wt_raw,
    float* __restrict__ out) {
  __shared__ __bf16 xl[18 * 18 * XSTR];    // halo tile [y18][x18][c], 22.0 KB
  __shared__ float  hxl[TW * TW * HSTR];   // conv result [pixel][h], 33.8 KB
  __shared__ float  s_inv[HID];
  __shared__ float  s_v[HID];
  __shared__ int    s_fl[NI];
  __shared__ float  s_bias[1];

  const __bf16* wt = (const __bf16*)wt_raw;
  const int tid = threadIdx.x;
  const int b   = blockIdx.z;
  const int gy0 = blockIdx.y * TW, gx0 = blockIdx.x * TW;

  // ---- stage x halo tile (f32 -> bf16), layout [y][x][c], coalesced reads
  for (int i = tid; i < CIN * 18 * 18; i += 256) {
    int c = i / 324, r = i - c * 324;
    int yy = r / 18, xx = r - yy * 18;
    int gy = gy0 + yy - 1, gx = gx0 + xx - 1;
    float v = 0.f;
    if ((unsigned)gy < IMG && (unsigned)gx < IMG)
      v = x[(((size_t)b * CIN + c) * IMG + gy) * IMG + gx];
    xl[(yy * 18 + xx) * XSTR + c] = (__bf16)v;
  }
  if (tid < HID) { s_inv[tid] = ws_inv[tid]; s_v[tid] = ws_v[tid]; }
  if (tid < NI)  s_fl[tid] = flags[b * NI + tid];
  if (tid == 0)  s_bias[0] = ws_bias[0];
  __syncthreads();

  // ---- WMMA: M = 256 pixels (16 row-tiles), N = 32 ch (2 tiles), K = 288
  const int wave  = tid >> 5;          // 8 waves, each: rows 2w, 2w+1
  const int lane  = tid & 31;
  const int mrow0 = wave * 2;
  const int mlane = lane & 15;         // A: M index; B: N column
  const int khalf = (lane >> 4) << 4;  // 0 or 16

  v8f acc00 = {}, acc01 = {}, acc10 = {}, acc11 = {};
  for (int tap = 0; tap < 9; ++tap) {
    const int dy = tap / 3, dx = tap - dy * 3;
    v16bf bf0, bf1, af0, af1;
    {
      const __bf16* p0 = wt + (size_t)(mlane)      * KTOT + tap * 32 + khalf;
      const __bf16* p1 = wt + (size_t)(16 + mlane) * KTOT + tap * 32 + khalf;
      const __bf16* q0 = &xl[((mrow0 + 0 + dy) * 18 + (mlane + dx)) * XSTR + khalf];
      const __bf16* q1 = &xl[((mrow0 + 1 + dy) * 18 + (mlane + dx)) * XSTR + khalf];
#pragma unroll
      for (int i = 0; i < 16; ++i) {
        bf0[i] = p0[i]; bf1[i] = p1[i];
        af0[i] = q0[i]; af1[i] = q1[i];
      }
    }
    acc00 = __builtin_amdgcn_wmma_f32_16x16x32_bf16(false, af0, false, bf0,
                                                    (short)0, acc00, false, false);
    acc01 = __builtin_amdgcn_wmma_f32_16x16x32_bf16(false, af0, false, bf1,
                                                    (short)0, acc01, false, false);
    acc10 = __builtin_amdgcn_wmma_f32_16x16x32_bf16(false, af1, false, bf0,
                                                    (short)0, acc10, false, false);
    acc11 = __builtin_amdgcn_wmma_f32_16x16x32_bf16(false, af1, false, bf1,
                                                    (short)0, acc11, false, false);
  }

  // ---- spill accumulators: D layout M = r + (lane>=16 ? 8 : 0), N = lane&15
  {
    const int mbase = (lane >= 16) ? 8 : 0;
    const int nch   = lane & 15;
#pragma unroll
    for (int r = 0; r < 8; ++r) {
      int xloc = r + mbase;
      hxl[((mrow0 + 0) * TW + xloc) * HSTR + nch]      = acc00[r];
      hxl[((mrow0 + 0) * TW + xloc) * HSTR + 16 + nch] = acc01[r];
      hxl[((mrow0 + 1) * TW + xloc) * HSTR + nch]      = acc10[r];
      hxl[((mrow0 + 1) * TW + xloc) * HSTR + 16 + nch] = acc11[r];
    }
  }
  __syncthreads();

  // ---- fused epilogue: one pixel per thread
  const int yloc = tid >> 4, xloc = tid & 15;
  const int gy = gy0 + yloc, gx = gx0 + xloc;
  const int yc = (gy == 0) ? 0 : ((gy == IMG - 1) ? 2 : 1);
  const int xc = (gx == 0) ? 0 : ((gx == IMG - 1) ? 2 : 1);
  const int cls = yc * 3 + xc;

  float hv[HID];
#pragma unroll
  for (int h = 0; h < HID; ++h) hv[h] = hxl[tid * HSTR + h] * s_inv[h];
  const float bias = s_bias[0];

  for (int n = 0; n < NI; ++n) {
    const float* ap = ws_a + (size_t)(n * HID) * 9 + cls;  // tiny, L0-resident
    float s = 0.f;
#pragma unroll
    for (int h = 0; h < HID; ++h) {
      float t = hv[h] + ap[h * 9];
      s += s_v[h] * ((t >= 0.f) ? t : LEAKY * t);
    }
    s += bias;
    if (!s_fl[n]) s = 0.f;
    out[(((size_t)b * NI + n) * IMG + gy) * IMG + gx] = s;
  }
}

// ---------------------------------------------------------------------------
extern "C" void kernel_launch(void* const* d_in, const int* in_sizes, int n_in,
                              void* d_out, int out_size, void* d_ws, size_t ws_size,
                              hipStream_t stream) {
  const float* x        = (const float*)d_in[0];
  const int*   masks    = (const int*)  d_in[1];
  const float* emb_w    = (const float*)d_in[2];
  const float* klin_w   = (const float*)d_in[3];
  const float* conv1_w  = (const float*)d_in[4];
  const float* bn_gamma = (const float*)d_in[5];
  const float* bn_beta  = (const float*)d_in[6];
  const float* bn_mean  = (const float*)d_in[7];
  const float* bn_var   = (const float*)d_in[8];
  const float* conv2_w  = (const float*)d_in[9];
  const float* conv2_b  = (const float*)d_in[10];
  float* out = (float*)d_out;

  // workspace layout
  float* ws_a    = (float*)d_ws;            // 2880 f32
  float* ws_inv  = ws_a + NI * HID * 9;     // 32
  float* ws_v    = ws_inv + HID;            // 32
  float* ws_bias = ws_v + HID;              // 1 (+7 pad)
  int*   ws_fl   = (int*)(ws_bias + 8);     // 40
  unsigned short* ws_wt = (unsigned short*)(ws_fl + 40);  // 32*288 bf16

  setup_kernel<<<1, 256, 0, stream>>>(emb_w, klin_w, conv1_w, bn_gamma, bn_beta,
                                      bn_mean, bn_var, conv2_w, conv2_b,
                                      ws_a, ws_inv, ws_v, ws_bias, ws_wt);
  active_kernel<<<NB * NI, 256, 0, stream>>>(masks, ws_fl);
  dim3 grid(IMG / TW, IMG / TW, NB);
  matte_main<<<grid, 256, 0, stream>>>(x, ws_a, ws_inv, ws_v, ws_bias, ws_fl,
                                       ws_wt, out);
}
